// SiFinder_59837484368168
// MI455X (gfx1250) — matrix operand order repaired and checked by reading, not compiled, not run on hardware
//
#include <hip/hip_runtime.h>

// ---------------- constants ----------------
namespace {
constexpr int KK  = 24;
constexpr int HH  = 480;
constexpr int WWi = 480;
constexpr int CCh = 3;
constexpr int PS  = CCh * KK * KK;     // 1728
constexpr int HC  = HH - KK + 1;       // 457
constexpr int NPATCH = 400;
constexpr int KSTEPS = PS / 32;        // 54
constexpr int NPIX = CCh * HH * WWi;   // 691200
constexpr float SIG = 1e-6f;

// workspace offsets (bytes), all >= 8B aligned
constexpr size_t O_Y16 = 0;            // f16 copy of y_dec: 1,382,400 B
constexpr size_t O_BP  = 1382400;      // packed patch B-matrix (f16): 1,382,400 B
constexpr size_t O_MX  = 2764800;      // mean_x[400]
constexpr size_t O_DX  = 2766848;      // |den_x|[400]
constexpr size_t O_TB  = 2768896;      // argmax table u64[400]
constexpr size_t O_HS  = 2772992;      // horizontal box sum: 3*480*457 f32
constexpr size_t O_HS2 = 5405696;      // horizontal box sum of squares
constexpr size_t O_SY  = 8038400;      // sum_y per position: 457*457 f32
constexpr size_t O_DNY = 8873984;      // |den_y| per position
}

typedef _Float16 v16h __attribute__((ext_vector_type(16)));
typedef float    v8f  __attribute__((ext_vector_type(8)));
typedef unsigned int u32x4 __attribute__((ext_vector_type(4)));
typedef unsigned int u32x8 __attribute__((ext_vector_type(8)));

__device__ inline v16h mk16h(u32x4 lo, u32x4 hi) {
    union { u32x8 u; v16h h; } t;
    t.u[0]=lo[0]; t.u[1]=lo[1]; t.u[2]=lo[2]; t.u[3]=lo[3];
    t.u[4]=hi[0]; t.u[5]=hi[1]; t.u[6]=hi[2]; t.u[7]=hi[3];
    return t.h;
}

// ---------------- prep: y_dec -> f16 ----------------
__global__ void k_y16(const float* __restrict__ y, _Float16* __restrict__ y16) {
    int i = blockIdx.x * 256 + threadIdx.x;
    if (i < NPIX) y16[i] = (_Float16)y[i];
}

// ---------------- prep: patch stats + WMMA-order pack ----------------
// Pack layout: element (patch p, k) -> bp[((g*54+t)*32 + lane)*16 + (kr&15)]
// with g=p/16, np=p%16, t=k/32, kr=k%32, lane = np + (kr&16).
__global__ void k_pack(const float* __restrict__ x, _Float16* __restrict__ bp,
                       float* __restrict__ mx, float* __restrict__ dxs,
                       unsigned long long* __restrict__ tb) {
    int p = blockIdx.x, tid = threadIdx.x;
    int pr = p / 20, pc = p % 20;
    float s = 0.f, s2 = 0.f;
    for (int k = tid; k < PS; k += 256) {
        int c = k / 576, rem = k - c * 576;
        int dy = rem / 24, dx = rem - dy * 24;
        float v = x[(c * HH + pr * KK + dy) * WWi + pc * KK + dx];
        s += v; s2 += v * v;
        int g = p >> 4, np = p & 15, t = k >> 5, kr = k & 31;
        int lane = np + (kr & 16);
        bp[((size_t)(g * KSTEPS + t) * 32 + lane) * 16 + (kr & 15)] = (_Float16)v;
    }
    __shared__ float sh[256], sh2[256];
    sh[tid] = s; sh2[tid] = s2; __syncthreads();
    for (int o = 128; o > 0; o >>= 1) {
        if (tid < o) { sh[tid] += sh[tid + o]; sh2[tid] += sh2[tid + o]; }
        __syncthreads();
    }
    if (tid == 0) {
        float m = sh[0] / (float)PS;
        mx[p] = m;
        dxs[p] = fabsf(sh2[0] / (float)PS - m * m);
        tb[p] = 0ull;                       // init argmax table
    }
}

// ---------------- prep: horizontal box sums (width 24) ----------------
__global__ void k_hs(const float* __restrict__ y, float* __restrict__ hs,
                     float* __restrict__ hs2) {
    int idx = blockIdx.x * 256 + threadIdx.x;
    if (idx >= CCh * HH * HC) return;
    int j = idx % HC, cy = idx / HC;
    const float* row = y + (size_t)cy * WWi + j;
    float s = 0.f, s2 = 0.f;
    for (int d = 0; d < KK; ++d) { float v = row[d]; s += v; s2 += v * v; }
    hs[idx] = s; hs2[idx] = s2;
}

// ---------------- prep: vertical+channel sums -> sum_y, |den_y| ----------------
__global__ void k_sv(const float* __restrict__ hs, const float* __restrict__ hs2,
                     float* __restrict__ sy, float* __restrict__ dny) {
    int idx = blockIdx.x * 256 + threadIdx.x;
    if (idx >= HC * HC) return;
    int i = idx / HC, j = idx % HC;
    float s = 0.f, s2 = 0.f;
    for (int c = 0; c < CCh; ++c)
        for (int d = 0; d < KK; ++d) {
            int r = (c * HH + i + d) * HC + j;
            s += hs[r]; s2 += hs2[r];
        }
    sy[idx] = s;
    const float inv = 1.f / (float)PS;
    dny[idx] = fabsf(s2 * inv - s * s * inv * inv);
}

// ---------------- main: implicit-GEMM correlation via WMMA f16 ----------------
struct AddrT { unsigned long long aA, aB, bb; };

__device__ __forceinline__ AddrT mkaddr(int t, int half, int i, int jc, int pg, int lane,
                                        unsigned long long ybase, unsigned long long bbase) {
    // A fragment: lanes 0-15 need K=32t+[0..7]/[8..15], lanes 16-31 K=32t+[16..23]/[24..31].
    // Every aligned 8-k group lies within one (c,dy) row (24 % 8 == 0) -> contiguous 16B.
    int k0a = t * 32 + (half << 4);
    int k0b = k0a + 8;
    int c0 = k0a / 576, r0 = k0a - c0 * 576, dy0 = r0 / 24, dx0 = r0 - dy0 * 24;
    int c1 = k0b / 576, r1 = k0b - c1 * 576, dy1 = r1 / 24, dx1 = r1 - dy1 * 24;
    AddrT a;
    a.aA = ybase + 2ull * (unsigned)((c0 * HH + i + dy0) * WWi + jc + dx0);
    a.aB = ybase + 2ull * (unsigned)((c1 * HH + i + dy1) * WWi + jc + dx1);
    a.bb = bbase + 2ull * (unsigned)((((size_t)(pg * 5) * KSTEPS + t) * 32 + lane) * 16);
    return a;
}

// Declare one fragment buffer set with prefix P.
#define DECLBUF(P) u32x4 P##a0, P##a1, P##b00, P##b01, P##b10, P##b11,                   \
                         P##b20, P##b21, P##b30, P##b31, P##b40, P##b41;

// One clause of 12 b128 loads: A-frag halves + 5 B subtiles (stride 55296 B).
#define ISSUE(T, P)                                                                      \
    { AddrT ad_ = mkaddr((T), half, i, jc, pg, lane, ybase, bbase);                      \
      asm volatile("s_clause 0xB\n\t"                                                    \
                   "global_load_b128 %0, %12, off\n\t"                                   \
                   "global_load_b128 %1, %13, off\n\t"                                   \
                   "global_load_b128 %2, %14, off\n\t"                                   \
                   "global_load_b128 %3, %14, off offset:16\n\t"                         \
                   "global_load_b128 %4, %14, off offset:55296\n\t"                      \
                   "global_load_b128 %5, %14, off offset:55312\n\t"                      \
                   "global_load_b128 %6, %14, off offset:110592\n\t"                     \
                   "global_load_b128 %7, %14, off offset:110608\n\t"                     \
                   "global_load_b128 %8, %14, off offset:165888\n\t"                     \
                   "global_load_b128 %9, %14, off offset:165904\n\t"                     \
                   "global_load_b128 %10, %14, off offset:221184\n\t"                    \
                   "global_load_b128 %11, %14, off offset:221200"                        \
                   : "=&v"(P##a0), "=&v"(P##a1), "=&v"(P##b00), "=&v"(P##b01),           \
                     "=&v"(P##b10), "=&v"(P##b11), "=&v"(P##b20), "=&v"(P##b21),         \
                     "=&v"(P##b30), "=&v"(P##b31), "=&v"(P##b40), "=&v"(P##b41)          \
                   : "v"(ad_.aA), "v"(ad_.aB), "v"(ad_.bb)); }

// Wait until the PREVIOUS iteration's 12 loads landed (12 newer ones still in flight).
#define WAITN(P, CNT)                                                                    \
    asm volatile("s_wait_loadcnt " CNT                                                   \
                 : "+v"(P##a0), "+v"(P##a1), "+v"(P##b00), "+v"(P##b01), "+v"(P##b10),   \
                   "+v"(P##b11), "+v"(P##b20), "+v"(P##b21), "+v"(P##b30), "+v"(P##b31), \
                   "+v"(P##b40), "+v"(P##b41));

#define WMMA5(P)                                                                         \
    { v16h Af_ = mk16h(P##a0, P##a1);                                                    \
      acc0 = __builtin_amdgcn_wmma_f32_16x16x32_f16(false, Af_, false,                   \
                 mk16h(P##b00, P##b01), (short)0, acc0, false, false);                   \
      acc1 = __builtin_amdgcn_wmma_f32_16x16x32_f16(false, Af_, false,                   \
                 mk16h(P##b10, P##b11), (short)0, acc1, false, false);                   \
      acc2 = __builtin_amdgcn_wmma_f32_16x16x32_f16(false, Af_, false,                   \
                 mk16h(P##b20, P##b21), (short)0, acc2, false, false);                   \
      acc3 = __builtin_amdgcn_wmma_f32_16x16x32_f16(false, Af_, false,                   \
                 mk16h(P##b30, P##b31), (short)0, acc3, false, false);                   \
      acc4 = __builtin_amdgcn_wmma_f32_16x16x32_f16(false, Af_, false,                   \
                 mk16h(P##b40, P##b41), (short)0, acc4, false, false); }

// Block: 8 waves. Wave w: positions (row i, cols j0w..j0w+15), all 80 patches of
// patch-group pg (5 subtiles of 16). K loop: 54 steps of 32, double-buffered.
__launch_bounds__(256, 1)
__global__ void k_gemm(const _Float16* __restrict__ y16, const _Float16* __restrict__ bp,
                       const float* __restrict__ mx, const float* __restrict__ dxs,
                       const float* __restrict__ sy, const float* __restrict__ dny,
                       unsigned long long* __restrict__ tb) {
    const int lane = threadIdx.x & 31;
    const int wave = threadIdx.x >> 5;
    const int half = lane >> 4;
    const int m16  = lane & 15;
    const int i    = blockIdx.y;                       // output row 0..456
    const int j0w  = blockIdx.x * 128 + wave * 16;     // wave's column base
    const int pg   = blockIdx.z;                       // patch group 0..4 (80 each)
    const int jc   = (j0w + m16 < HC) ? (j0w + m16) : (HC - 1);  // clamp for loads

    v8f acc0{}, acc1{}, acc2{}, acc3{}, acc4{};
    const unsigned long long ybase = (unsigned long long)y16;
    const unsigned long long bbase = (unsigned long long)bp;

    DECLBUF(x)
    DECLBUF(y)

    // Software pipeline: prefetch t+1 while multiplying t.
    ISSUE(0, x)
    for (int t = 0; t < KSTEPS - 2; t += 2) {
        ISSUE(t + 1, y)
        WAITN(x, "0xc")
        WMMA5(x)
        ISSUE(t + 2, x)
        WAITN(y, "0xc")
        WMMA5(y)
    }
    // tail: t = 52, 53
    ISSUE(KSTEPS - 1, y)
    WAITN(x, "0xc")
    WMMA5(x)
    WAITN(y, "0x0")
    WMMA5(y)

    // Epilogue. C layout: VGPR r, lane l -> M = r + 8*(l>>4), N = l&15.
    float syv[8], dyv[8];
    for (int r = 0; r < 8; ++r) {
        int jj = j0w + r + 8 * half;
        int pos = i * HC + ((jj < HC) ? jj : (HC - 1));
        syv[r] = sy[pos]; dyv[r] = dny[pos];
    }
    const float inv = 1.f / (float)PS;
    v8f accs[5] = {acc0, acc1, acc2, acc3, acc4};
    for (int s = 0; s < 5; ++s) {
        int n = pg * 80 + s * 16 + m16;
        float mxv = mx[n], dxv = dxs[n];
        float best = -1e30f; int bidx = 0;
        for (int r = 0; r < 8; ++r) {
            int jj = j0w + r + 8 * half;
            float corr = (2.f * (accs[s][r] - mxv * syv[r]) * inv + SIG)
                         / (dxv + dyv[r] + SIG);
            if (jj >= HC) corr = -1e30f;
            if (corr > best) { best = corr; bidx = i * HC + jj; }
        }
        float ob = __shfl_xor(best, 16, 32);
        int   oi = __shfl_xor(bidx, 16, 32);
        if (ob > best || (ob == best && oi < bidx)) { best = ob; bidx = oi; }
        if (half == 0 && best > -1e29f) {
            unsigned u = __float_as_uint(best);
            u = (u & 0x80000000u) ? ~u : (u | 0x80000000u);
            unsigned long long key =
                ((unsigned long long)u << 32) | (unsigned)(~bidx);  // ~idx: ties -> lowest idx
            atomicMax(&tb[n], key);
        }
    }
}

// ---------------- final: gather winning KxK patches from y, fold to image ----------------
__global__ void k_gather(const float* __restrict__ y, const unsigned long long* __restrict__ tb,
                         float* __restrict__ out) {
    int idx = blockIdx.x * 256 + threadIdx.x;
    if (idx >= NPIX) return;
    int c = idx / (HH * WWi);
    int rem = idx - c * HH * WWi;
    int h = rem / WWi, w = rem - h * WWi;
    int p = (h / KK) * 20 + (w / KK);
    unsigned pos = ~(unsigned)(tb[p] & 0xFFFFFFFFull);
    int rr = pos / HC, cc = pos - rr * HC;
    out[idx] = y[(c * HH + rr + h % KK) * WWi + cc + w % KK];
}

// ---------------- launch ----------------
extern "C" void kernel_launch(void* const* d_in, const int* in_sizes, int n_in,
                              void* d_out, int out_size, void* d_ws, size_t ws_size,
                              hipStream_t stream) {
    (void)in_sizes; (void)n_in; (void)out_size; (void)ws_size;
    const float* x_dec = (const float*)d_in[0];
    const float* y_dec = (const float*)d_in[1];
    const float* y_img = (const float*)d_in[2];
    float* out = (float*)d_out;
    char* ws = (char*)d_ws;

    _Float16* y16 = (_Float16*)(ws + O_Y16);
    _Float16* bpk = (_Float16*)(ws + O_BP);
    float* mxp  = (float*)(ws + O_MX);
    float* dxp  = (float*)(ws + O_DX);
    unsigned long long* tbp = (unsigned long long*)(ws + O_TB);
    float* hs  = (float*)(ws + O_HS);
    float* hs2 = (float*)(ws + O_HS2);
    float* syp = (float*)(ws + O_SY);
    float* dnp = (float*)(ws + O_DNY);

    k_y16 <<<(NPIX + 255) / 256, 256, 0, stream>>>(y_dec, y16);
    k_pack<<<NPATCH, 256, 0, stream>>>(x_dec, bpk, mxp, dxp, tbp);
    k_hs  <<<(CCh * HH * HC + 255) / 256, 256, 0, stream>>>(y_dec, hs, hs2);
    k_sv  <<<(HC * HC + 255) / 256, 256, 0, stream>>>(hs, hs2, syp, dnp);
    k_gemm<<<dim3(4, HC, 5), 256, 0, stream>>>(y16, bpk, mxp, dxp, syp, dnp, tbp);
    k_gather<<<(NPIX + 255) / 256, 256, 0, stream>>>(y_img, tbp, out);
}
